// ZincGNN_82927228551355
// MI455X (gfx1250) — compile-verified, matrix operand order checked by reading, not run on hardware
//
#include <hip/hip_runtime.h>
#include <hip/hip_bf16.h>

#define N_NODES  150000
#define N_EDGES  300000
#define N_GRAPHS 6000
#define HID      256

typedef __attribute__((ext_vector_type(16))) __bf16 v16bf;
typedef __attribute__((ext_vector_type(8)))  float  v8f;

// ---------------- elementwise / scatter kernels ----------------

__global__ void k_zero_f32(float* __restrict__ d, int n) {
  int i = blockIdx.x * blockDim.x + threadIdx.x;
  if (i < n) d[i] = 0.0f;
}

__global__ void k_copy_f32(const float* __restrict__ s, float* __restrict__ d, int n) {
  int i = blockIdx.x * blockDim.x + threadIdx.x;
  if (i < n) d[i] = s[i];
}

// Copy x (N_NODES x 7, packed) into agg (N_NODES x 32, zero-padded rows)
__global__ void k_copy_pad7(const float* __restrict__ x, float* __restrict__ agg, int nNodes) {
  int i = blockIdx.x * blockDim.x + threadIdx.x;
  if (i >= nNodes * 7) return;
  int node = i / 7;
  int c = i - node * 7;
  agg[(size_t)node * 32 + c] = x[i];
}

// agg[dst[e]*ldAgg + c] += x[src[e]*ldX + c]
__global__ void k_scatter_add(const float* __restrict__ x, int ldX,
                              const int* __restrict__ src,
                              const int* __restrict__ dst,
                              float* __restrict__ agg, int ldAgg,
                              int nE, int F) {
  int i = blockIdx.x * blockDim.x + threadIdx.x;
  int total = nE * F;
  if (i >= total) return;
  int e = i / F;
  int c = i - e * F;
  int s = src[e];
  int d = dst[e];
  atomicAdd(&agg[(size_t)d * ldAgg + c], x[(size_t)s * ldX + c]);
}

__global__ void k_pool_sum(const float* __restrict__ h,
                           const int* __restrict__ batch,
                           float* __restrict__ sums, int nNodes) {
  int i = blockIdx.x * blockDim.x + threadIdx.x;
  if (i >= nNodes * HID) return;
  int node = i >> 8;          // HID == 256
  int c    = i & 255;
  int g = batch[node];
  atomicAdd(&sums[(size_t)g * HID + c], h[i]);
}

__global__ void k_pool_count(const int* __restrict__ batch,
                             float* __restrict__ counts, int nNodes) {
  int i = blockIdx.x * blockDim.x + threadIdx.x;
  if (i < nNodes) atomicAdd(&counts[batch[i]], 1.0f);
}

__global__ void k_pool_div(float* __restrict__ sums,
                           const float* __restrict__ counts, int nG) {
  int i = blockIdx.x * blockDim.x + threadIdx.x;
  if (i >= nG * HID) return;
  int g = i >> 8;
  sums[i] = sums[i] / fmaxf(counts[g], 1.0f);
}

// out[g][j] = bh2[j] + sum_c hm[g][c] * Wh2[c][j]   (K=128, N=3 — too thin for WMMA)
__global__ void k_head2(const float* __restrict__ hm,
                        const float* __restrict__ W2,
                        const float* __restrict__ b2,
                        float* __restrict__ out, int G) {
  int i = blockIdx.x * blockDim.x + threadIdx.x;
  if (i >= G * 3) return;
  int g = i / 3;
  int j = i - g * 3;
  float s = b2[j];
#pragma unroll 4
  for (int c = 0; c < 128; ++c) s += hm[(size_t)g * 128 + c] * W2[c * 3 + j];
  out[i] = s;
}

// ---------------- weight pre-pack into WMMA fragment-native layout ----------
// For each 32x16 (K x N) tile, store 32 lanes x 16 bf16 contiguous so a lane's
// whole B fragment is one 32-byte load.  lane l, elem e -> k = base + khalf(l)
// + (e>>3)*16 + (e&7), n = nt*16 + (l&15).  Kp is K padded to a multiple of 32
// (rows >= K zero-filled).
__global__ void k_pack_w(const float* __restrict__ W, int K, int N, int Kp,
                         __bf16* __restrict__ out) {
  int i = blockIdx.x * blockDim.x + threadIdx.x;
  if (i >= Kp * N) return;
  int e    = i & 15;
  int lane = (i >> 4) & 31;
  int blk  = i >> 9;                  // kt * (N/16) + nt
  int ntiles = N >> 4;
  int nt = blk % ntiles;
  int kt = blk / ntiles;
  int k = kt * 32 + ((lane & 16) ? 8 : 0) + ((e >> 3) << 4) + (e & 7);
  int n = nt * 16 + (lane & 15);
  float v = (k < K) ? W[(size_t)k * N + n] : 0.0f;
  out[i] = (__bf16)v;
}

// ---------------- WMMA GEMM: C = act(A(MxK) * W(KxN) + bias) ----------------
// Block = 128 threads = 4 waves; block tile = 128x64; wave tile = 32x64
// (2 A fragments x 4 B fragments -> 8 WMMAs per 32-wide K step).
// A fragments: direct fp32 global loads (row index clamped to M-1: a garbage
// A row m only contributes to D row m, which is never stored when m >= M),
// converted to bf16 in registers.  B fragments: contiguous v16bf loads from
// the pre-packed weight buffer (L2-resident), hoisted ahead of the WMMA burst.
// No LDS, no barriers.  Requires: K % 32 == 0, lda 32B-aligned, N % 16 == 0.
__global__ __launch_bounds__(128)
void k_gemm_wmma(const float* __restrict__ A, int lda,
                 const __bf16* __restrict__ packedB,
                 const float* __restrict__ bias,
                 float* __restrict__ C, int ldc,
                 int M, int N, int K, int doRelu)
{
  const int tid    = threadIdx.x;
  const int wid    = tid >> 5;          // wave id 0..3 (wave32)
  const int lane   = tid & 31;
  const int mBase  = blockIdx.x * 128 + wid * 32;
  const int ntBase = blockIdx.y * 4;    // 4 column tiles (64 cols) per block
  const int ntiles = N >> 4;
  const int khalf  = (lane & 16) ? 8 : 0;

  // Two 16-row A sub-tiles per wave; clamp rows so OOB lanes read row M-1.
  int row0 = mBase + (lane & 15);
  int row1 = row0 + 16;
  if (row0 > M - 1) row0 = M - 1;
  if (row1 > M - 1) row1 = M - 1;
  const float* ap0 = A + (size_t)row0 * lda + khalf;
  const float* ap1 = A + (size_t)row1 * lda + khalf;
  const v16bf* __restrict__ pB = (const v16bf*)packedB;

  v8f acc[2][4];
#pragma unroll
  for (int m = 0; m < 2; ++m)
#pragma unroll
    for (int t = 0; t < 4; ++t) acc[m][t] = (v8f){};

  const int K32 = K >> 5;
  for (int kt32 = 0; kt32 < K32; ++kt32) {
    const int kb = kt32 << 5;

    // ---- B fragments first: 8 contiguous b128 loads, clause-friendly
    v16bf b[4];
    const size_t bbase = ((size_t)kt32 * ntiles + ntBase) * 32 + lane;
#pragma unroll
    for (int nt = 0; nt < 4; ++nt) b[nt] = pB[bbase + (size_t)nt * 32];

    // ---- A fragments: k in [kb+khalf, +8) u [kb+khalf+16, +24) per row
    v16bf a[2];
#pragma unroll
    for (int m = 0; m < 2; ++m) {
      const float* ap = (m == 0 ? ap0 : ap1) + kb;
      float4 q0 = *(const float4*)(ap);
      float4 q1 = *(const float4*)(ap + 4);
      float4 q2 = *(const float4*)(ap + 16);
      float4 q3 = *(const float4*)(ap + 20);
      a[m][0]  = (__bf16)q0.x; a[m][1]  = (__bf16)q0.y;
      a[m][2]  = (__bf16)q0.z; a[m][3]  = (__bf16)q0.w;
      a[m][4]  = (__bf16)q1.x; a[m][5]  = (__bf16)q1.y;
      a[m][6]  = (__bf16)q1.z; a[m][7]  = (__bf16)q1.w;
      a[m][8]  = (__bf16)q2.x; a[m][9]  = (__bf16)q2.y;
      a[m][10] = (__bf16)q2.z; a[m][11] = (__bf16)q2.w;
      a[m][12] = (__bf16)q3.x; a[m][13] = (__bf16)q3.y;
      a[m][14] = (__bf16)q3.z; a[m][15] = (__bf16)q3.w;
    }

    // ---- 8 back-to-back WMMAs (B reused across the two row tiles)
#pragma unroll
    for (int nt = 0; nt < 4; ++nt) {
#pragma unroll
      for (int m = 0; m < 2; ++m) {
        // (neg_a, A, neg_b, B, c_mod, C, reuse_a, reuse_b)
        acc[m][nt] = __builtin_amdgcn_wmma_f32_16x16x32_bf16(
            false, a[m], false, b[nt], (short)0, acc[m][nt], false, false);
      }
    }
  }

  // Epilogue: D layout — VGPR r holds M=r (lanes 0-15) / M=r+8 (lanes 16-31)
  const int rowBase = mBase + ((lane >> 4) << 3);
#pragma unroll
  for (int m = 0; m < 2; ++m) {
#pragma unroll
    for (int nt = 0; nt < 4; ++nt) {
      int col = (ntBase + nt) * 16 + (lane & 15);
      if (col < N) {
        float bv = bias[col];
#pragma unroll
        for (int r = 0; r < 8; ++r) {
          int rw = rowBase + m * 16 + r;
          if (rw < M) {
            float v = acc[m][nt][r] + bv;
            if (doRelu) v = fmaxf(v, 0.0f);
            C[(size_t)rw * ldc + col] = v;
          }
        }
      }
    }
  }
}

// ---------------- driver ----------------

extern "C" void kernel_launch(void* const* d_in, const int* in_sizes, int n_in,
                              void* d_out, int out_size, void* d_ws, size_t ws_size,
                              hipStream_t stream) {
  (void)in_sizes; (void)n_in; (void)out_size; (void)ws_size;

  const float* x     = (const float*)d_in[0];
  const int*   ei    = (const int*)d_in[1];
  const int*   src   = ei;              // edge_index row 0
  const int*   dst   = ei + N_EDGES;    // edge_index row 1
  const int*   batch = (const int*)d_in[2];
  const float* Wa[4] = {(const float*)d_in[3],  (const float*)d_in[7],
                        (const float*)d_in[11], (const float*)d_in[15]};
  const float* ba[4] = {(const float*)d_in[4],  (const float*)d_in[8],
                        (const float*)d_in[12], (const float*)d_in[16]};
  const float* Wb[4] = {(const float*)d_in[5],  (const float*)d_in[9],
                        (const float*)d_in[13], (const float*)d_in[17]};
  const float* bb[4] = {(const float*)d_in[6],  (const float*)d_in[10],
                        (const float*)d_in[14], (const float*)d_in[18]};
  const float* Wh1 = (const float*)d_in[19];
  const float* bh1 = (const float*)d_in[20];
  const float* Wh2 = (const float*)d_in[21];
  const float* bh2 = (const float*)d_in[22];
  float* out = (float*)d_out;

  // Workspace carve-up (256B aligned)
  char* ws = (char*)d_ws;
  size_t off = 0;
  auto carve = [&](size_t bytes) -> char* {
    char* p = ws + off;
    off += (bytes + 255) & ~(size_t)255;
    return p;
  };
  float*  feat   = (float*)carve((size_t)N_NODES * HID * 4);
  float*  agg    = (float*)carve((size_t)N_NODES * HID * 4);
  float*  mid    = (float*)carve((size_t)N_NODES * HID * 4);
  float*  sums   = (float*)carve((size_t)N_GRAPHS * HID * 4);
  float*  counts = (float*)carve((size_t)N_GRAPHS * 4);
  float*  hm     = (float*)carve((size_t)N_GRAPHS * 128 * 4);
  __bf16* pW1a   = (__bf16*)carve((size_t)32 * HID * 2);        // Kp=32, N=256
  __bf16* pWsq[7];                                              // seven 256x256
  for (int i = 0; i < 7; ++i) pWsq[i] = (__bf16*)carve((size_t)HID * HID * 2);
  __bf16* pWh1   = (__bf16*)carve((size_t)HID * 128 * 2);       // Kp=256, N=128

  const dim3 blk(256);
  const dim3 gemmBlk(128);
  const dim3 gridBig((N_NODES + 127) / 128, HID / 64);   // 1172 x 4

  // ---- Pre-pack all weights into fragment-native bf16 layout ----
  {
    int n1 = 32 * HID;
    k_pack_w<<<(n1 + 255) / 256, blk, 0, stream>>>(Wa[0], 7, HID, 32, pW1a);
    int ns = HID * HID;
    const float* sq[7] = {Wb[0], Wa[1], Wb[1], Wa[2], Wb[2], Wa[3], Wb[3]};
    for (int i = 0; i < 7; ++i)
      k_pack_w<<<(ns + 255) / 256, blk, 0, stream>>>(sq[i], HID, HID, HID, pWsq[i]);
    int nh = HID * 128;
    k_pack_w<<<(nh + 255) / 256, blk, 0, stream>>>(Wh1, HID, 128, HID, pWh1);
  }

  // ---- Layer 1 (F_in = 7, padded to lda = 32) ----
  {
    int n = N_NODES * 32;
    k_zero_f32<<<(n + 255) / 256, blk, 0, stream>>>(agg, n);
    int n7 = N_NODES * 7;
    k_copy_pad7<<<(n7 + 255) / 256, blk, 0, stream>>>(x, agg, N_NODES);
    int t = N_EDGES * 7;
    k_scatter_add<<<(t + 255) / 256, blk, 0, stream>>>(x, 7, src, dst, agg, 32, N_EDGES, 7);
    k_gemm_wmma<<<gridBig, gemmBlk, 0, stream>>>(agg, 32, pW1a, ba[0], mid, HID,
                                                 N_NODES, HID, 32, 1);
    k_gemm_wmma<<<gridBig, gemmBlk, 0, stream>>>(mid, HID, pWsq[0], bb[0], feat, HID,
                                                 N_NODES, HID, HID, 1);
  }

  // ---- Layers 2..4 ----
  for (int l = 1; l < 4; ++l) {
    int n = N_NODES * HID;
    k_copy_f32<<<(n + 255) / 256, blk, 0, stream>>>(feat, agg, n);
    int t = N_EDGES * HID;
    k_scatter_add<<<(t + 255) / 256, blk, 0, stream>>>(feat, HID, src, dst, agg, HID, N_EDGES, HID);
    k_gemm_wmma<<<gridBig, gemmBlk, 0, stream>>>(agg, HID, pWsq[2 * l - 1], ba[l], mid, HID,
                                                 N_NODES, HID, HID, 1);
    k_gemm_wmma<<<gridBig, gemmBlk, 0, stream>>>(mid, HID, pWsq[2 * l], bb[l], feat, HID,
                                                 N_NODES, HID, HID, 1);
  }

  // ---- Global mean pool ----
  {
    int n = N_GRAPHS * HID;
    k_zero_f32<<<(n + 255) / 256, blk, 0, stream>>>(sums, n);
    k_zero_f32<<<(N_GRAPHS + 255) / 256, blk, 0, stream>>>(counts, N_GRAPHS);
    int t = N_NODES * HID;
    k_pool_sum<<<(t + 255) / 256, blk, 0, stream>>>(feat, batch, sums, N_NODES);
    k_pool_count<<<(N_NODES + 255) / 256, blk, 0, stream>>>(batch, counts, N_NODES);
    k_pool_div<<<(n + 255) / 256, blk, 0, stream>>>(sums, counts, N_GRAPHS);
  }

  // ---- Head: relu(pooled @ Wh1 + bh1) @ Wh2 + bh2 ----
  {
    dim3 gh((N_GRAPHS + 127) / 128, 128 / 64);
    k_gemm_wmma<<<gh, gemmBlk, 0, stream>>>(sums, HID, pWh1, bh1, hm, 128,
                                            N_GRAPHS, 128, HID, 1);
    int t = N_GRAPHS * 3;
    k_head2<<<(t + 255) / 256, blk, 0, stream>>>(hm, Wh2, bh2, out, N_GRAPHS);
  }
}
